// RNN_73650099191877
// MI455X (gfx1250) — matrix operand order repaired and checked by reading, not compiled, not run on hardware
//
#include <hip/hip_runtime.h>
#include <hip/hip_bf16.h>
#include <math.h>

#define DIM_X 1024
#define DIM_Y 256
#define DIM_U 64
#define DIM_S 64
#define BATCH 256
#define STEPS 256

typedef __attribute__((ext_vector_type(16))) __bf16 v16bf;
typedef __attribute__((ext_vector_type(8)))  __bf16 v8bf;
typedef __attribute__((ext_vector_type(8)))  float  v8f;

#if __has_builtin(__builtin_amdgcn_tanhf)
#define TANHF(x) __builtin_amdgcn_tanhf(x)
#elif __has_builtin(__builtin_amdgcn_tanh_f32)
#define TANHF(x) __builtin_amdgcn_tanh_f32(x)
#else
#define TANHF(x) tanhf(x)
#endif

// ---------------------------------------------------------------------------
// Packed B-matrix tile layout (dense 32x16 16-bit B fragment, per ISA
// CDNA5 ISA): lane L (h=L/16, n=L%16) holds v[j] = B[k = 8*h + j (j<8), 16 + 8*h + j-8 (j>=8)][n].
// Tile (kt,nt) stored contiguously: dst[(kt*NT+nt)*512 + L*16 + j].
// B[k][n] = src[nglob * cols + kglob]   (since pre = x @ src^T)
// ---------------------------------------------------------------------------
__global__ __launch_bounds__(256) void pack_kernel(const float* __restrict__ W,
                                                   const float* __restrict__ Bs,
                                                   const float* __restrict__ Wy,
                                                   __bf16* __restrict__ ws) {
  int wave = threadIdx.x >> 5, lane = threadIdx.x & 31;
  int tile = blockIdx.x * 8 + wave;
  const float* src; __bf16* dst; int NT, cols, t0;
  if (tile < 2048)      { src = W;  dst = ws;           NT = 64; cols = 1024; t0 = tile;        } // W: 32x64 tiles
  else if (tile < 2176) { src = Bs; dst = ws + 1048576; NT = 64; cols = 64;   t0 = tile - 2048; } // Bs: 2x64 tiles
  else if (tile < 2688) { src = Wy; dst = ws + 1114112; NT = 16; cols = 1024; t0 = tile - 2176; } // Wy: 32x16 tiles
  else return;
  int kt = t0 / NT, nt = t0 % NT;
  int h = lane >> 4, n = nt * 16 + (lane & 15);
  const float* srow = src + (size_t)n * cols + kt * 32 + 8 * h;
  v16bf v;
#pragma unroll
  for (int j = 0; j < 16; ++j) v[j] = (__bf16)srow[j < 8 ? j : j + 8];
  *(v16bf*)(dst + (size_t)t0 * 512 + lane * 16) = v;
}

// x_seq[:,0,:] = x0 (fp32) and bf16 copy into the state mirror/history.
__global__ __launch_bounds__(256) void init_x_kernel(const float* __restrict__ x0,
                                                     float* __restrict__ x_seq,
                                                     __bf16* __restrict__ xbf,
                                                     int bf_row_stride) {
  int i = blockIdx.x * 256 + threadIdx.x;          // 0 .. 256*1024-1
  int b = i >> 10, x = i & 1023;
  float v = x0[i];
  x_seq[(size_t)b * STEPS * DIM_X + x] = v;        // x_seq[b, 0, x]
  xbf[(size_t)b * bf_row_stride + x] = (__bf16)v;
}

// A fragment (16x32 bf16, per ISA): lane L holds row M = L%16;
// v[0..7] = K kbase+8h+0..7, v[8..15] = K kbase+16+8h+0..7.  Pure b128 loads.
__device__ __forceinline__ v16bf load_a_bf(const __bf16* __restrict__ row,
                                           int kbase, int h) {
  v8bf lo = *(const v8bf*)(row + kbase + 8 * h);
  v8bf hi = *(const v8bf*)(row + kbase + 16 + 8 * h);
  v16bf a;
#pragma unroll
  for (int j = 0; j < 8; ++j) { a[j] = lo[j]; a[j + 8] = hi[j]; }
  return a;
}

// Same fragment built from fp32 memory (native v_cvt_pk_bf16_f32 casts).
__device__ __forceinline__ v16bf load_a_f32(const float* __restrict__ row,
                                            int kbase, int h) {
  const float* p0 = row + kbase + 8 * h;
  const float* p1 = p0 + 16;
  float4 f0 = *(const float4*)(p0);
  float4 f1 = *(const float4*)(p0 + 4);
  float4 f2 = *(const float4*)(p1);
  float4 f3 = *(const float4*)(p1 + 4);
  v16bf a;
  a[0]  = (__bf16)f0.x; a[1]  = (__bf16)f0.y; a[2]  = (__bf16)f0.z; a[3]  = (__bf16)f0.w;
  a[4]  = (__bf16)f1.x; a[5]  = (__bf16)f1.y; a[6]  = (__bf16)f1.z; a[7]  = (__bf16)f1.w;
  a[8]  = (__bf16)f2.x; a[9]  = (__bf16)f2.y; a[10] = (__bf16)f2.z; a[11] = (__bf16)f2.w;
  a[12] = (__bf16)f3.x; a[13] = (__bf16)f3.y; a[14] = (__bf16)f3.z; a[15] = (__bf16)f3.w;
  return a;
}

// One recurrence step: x_{t+1} = 0.9 x_t + 0.1 tanh(x_t W^T + s_t Bs^T + u_pad + b)
// Grid: 16 blocks x 256 threads = 128 waves; wave -> (mtile32 0..7, ngroup 0..15);
// each wave computes a 32(M) x 64(N) tile: 2 A-frags x 4 B-frags -> 8 WMMAs/k-step.
__global__ __launch_bounds__(256) void step_kernel(const __bf16* __restrict__ xbf_rd,
                                                   __bf16* __restrict__ xbf_wr,
                                                   int bf_row_stride,
                                                   const float* __restrict__ s_seq,
                                                   const float* __restrict__ u_seq,
                                                   const float* __restrict__ bias,
                                                   const __bf16* __restrict__ packW,
                                                   const __bf16* __restrict__ packBs,
                                                   float* __restrict__ x_seq,
                                                   int t) {
  int lane   = threadIdx.x & 31;
  int wid    = blockIdx.x * 8 + (threadIdx.x >> 5);
  int mtile  = wid >> 4;     // 0..7 (32 rows each)
  int ngroup = wid & 15;     // 0..15 (64 cols each)
  int h = lane >> 4, lm = lane & 15;

  int m0 = mtile * 32 + lm, m1 = m0 + 16;
  const __bf16* arow0 = xbf_rd + (size_t)m0 * bf_row_stride;
  const __bf16* arow1 = xbf_rd + (size_t)m1 * bf_row_stride;
  const float*  srow0 = s_seq + ((size_t)m0 * STEPS + t) * DIM_S;
  const float*  srow1 = s_seq + ((size_t)m1 * STEPS + t) * DIM_S;
  const __bf16* wbase = packW + (size_t)(ngroup * 4) * 512 + lane * 16;

  // Warm the B stream (wave-uniform, outside the loop: keeps the K-loop CFG flat).
  __builtin_prefetch(wbase, 0, 3);
  __builtin_prefetch(wbase + 64 * 512, 0, 3);

  v8f acc[2][4] = {};

  // K = 1024 over W (32 k-steps); branch-free body: pure loads + 8 WMMAs.
  for (int kt = 0; kt < 32; ++kt) {
    v16bf a0 = load_a_bf(arow0, kt * 32, h);
    v16bf a1 = load_a_bf(arow1, kt * 32, h);
#pragma unroll
    for (int q = 0; q < 4; ++q) {
      v16bf b = *(const v16bf*)(wbase + (size_t)(kt * 64 + q) * 512);
      acc[0][q] = __builtin_amdgcn_wmma_f32_16x16x32_bf16(false, a0, false, b,
                                                          (short)0, acc[0][q], false, false);
      acc[1][q] = __builtin_amdgcn_wmma_f32_16x16x32_bf16(false, a1, false, b,
                                                          (short)0, acc[1][q], false, false);
    }
  }
  // K-extension: + s_t @ Bs^T (K = 64, 2 k-steps)
#pragma unroll
  for (int kt = 0; kt < 2; ++kt) {
    v16bf a0 = load_a_f32(srow0, kt * 32, h);
    v16bf a1 = load_a_f32(srow1, kt * 32, h);
#pragma unroll
    for (int q = 0; q < 4; ++q) {
      int nt = ngroup * 4 + q;
      v16bf b = *(const v16bf*)(packBs + (size_t)(kt * 64 + nt) * 512 + lane * 16);
      acc[0][q] = __builtin_amdgcn_wmma_f32_16x16x32_bf16(false, a0, false, b,
                                                          (short)0, acc[0][q], false, false);
      acc[1][q] = __builtin_amdgcn_wmma_f32_16x16x32_bf16(false, a1, false, b,
                                                          (short)0, acc[1][q], false, false);
    }
  }

  // Epilogue: + u_pad + bias, tanh, leak blend; write fp32 x_{t+1} and bf16 state.
  // C/D layout: lane half h, reg r -> M = base + 8h + r, N = lm.
  const float A = 0.1f;
#pragma unroll
  for (int mi = 0; mi < 2; ++mi) {
#pragma unroll
    for (int q = 0; q < 4; ++q) {
      int ncol = (ngroup * 4 + q) * 16 + lm;
      float bv = bias[ncol];
#pragma unroll
      for (int r = 0; r < 8; ++r) {
        int m = mtile * 32 + mi * 16 + 8 * h + r;
        float pre = acc[mi][q][r] + bv;
        if (ncol < DIM_U)
          pre += u_seq[((size_t)m * STEPS + t) * DIM_U + ncol];
        size_t xo = ((size_t)m * STEPS + t) * DIM_X + ncol;
        float xp = x_seq[xo];
        float xn = (1.0f - A) * xp + A * TANHF(pre);
        x_seq[xo + DIM_X] = xn;
        xbf_wr[(size_t)m * bf_row_stride + ncol] = (__bf16)xn;
      }
    }
  }
}

// y = x_seq @ Wy^T + by : M = B*T = 65536, K = 1024, N = 256.
// Grid: 1024 blocks x 256 threads = 8192 waves; wave -> (mtile32 0..2047, ngroup 0..3).
template <bool USE_HIST>
__global__ __launch_bounds__(256) void y_kernel(const float* __restrict__ x_seq,
                                                const __bf16* __restrict__ xhist,
                                                const __bf16* __restrict__ packWy,
                                                const float* __restrict__ by,
                                                float* __restrict__ y) {
  int lane   = threadIdx.x & 31;
  int wid    = blockIdx.x * 8 + (threadIdx.x >> 5);
  int mtile  = wid >> 2;     // 0..2047
  int ngroup = wid & 3;      // 0..3
  int h = lane >> 4, lm = lane & 15;

  int m0 = mtile * 32 + lm, m1 = m0 + 16;         // flat rows b*STEPS + t
  const __bf16* brow0 = xhist + (size_t)m0 * DIM_X;
  const __bf16* brow1 = xhist + (size_t)m1 * DIM_X;
  const float*  frow0 = x_seq + (size_t)m0 * DIM_X;
  const float*  frow1 = x_seq + (size_t)m1 * DIM_X;
  const __bf16* wbase = packWy + (size_t)(ngroup * 4) * 512 + lane * 16;

  __builtin_prefetch(wbase, 0, 3);

  v8f acc[2][4] = {};
  for (int kt = 0; kt < 32; ++kt) {
    v16bf a0 = USE_HIST ? load_a_bf(brow0, kt * 32, h) : load_a_f32(frow0, kt * 32, h);
    v16bf a1 = USE_HIST ? load_a_bf(brow1, kt * 32, h) : load_a_f32(frow1, kt * 32, h);
#pragma unroll
    for (int q = 0; q < 4; ++q) {
      v16bf b = *(const v16bf*)(wbase + (size_t)(kt * 16 + q) * 512);
      acc[0][q] = __builtin_amdgcn_wmma_f32_16x16x32_bf16(false, a0, false, b,
                                                          (short)0, acc[0][q], false, false);
      acc[1][q] = __builtin_amdgcn_wmma_f32_16x16x32_bf16(false, a1, false, b,
                                                          (short)0, acc[1][q], false, false);
    }
  }
#pragma unroll
  for (int mi = 0; mi < 2; ++mi) {
#pragma unroll
    for (int q = 0; q < 4; ++q) {
      int ncol = (ngroup * 4 + q) * 16 + lm;
      float bv = by[ncol];
#pragma unroll
      for (int r = 0; r < 8; ++r) {
        int m = mtile * 32 + mi * 16 + 8 * h + r;
        y[(size_t)m * DIM_Y + ncol] = acc[mi][q][r] + bv;
      }
    }
  }
}

extern "C" void kernel_launch(void* const* d_in, const int* in_sizes, int n_in,
                              void* d_out, int out_size, void* d_ws, size_t ws_size,
                              hipStream_t stream) {
  const float* x0    = (const float*)d_in[0];
  const float* s_seq = (const float*)d_in[1];
  const float* u_seq = (const float*)d_in[2];
  const float* Bs    = (const float*)d_in[3];
  const float* W     = (const float*)d_in[4];
  const float* b     = (const float*)d_in[5];
  const float* Wy    = (const float*)d_in[6];
  const float* by    = (const float*)d_in[7];

  float* x_seq = (float*)d_out;                               // [B, T, X]
  float* y_seq = x_seq + (size_t)BATCH * STEPS * DIM_X;       // [B, T, Y]

  __bf16* ws     = (__bf16*)d_ws;
  __bf16* packW  = ws;                 // 32*64*512 = 1,048,576 bf16
  __bf16* packBs = ws + 1048576;       //  2*64*512 =    65,536 bf16
  __bf16* packWy = ws + 1114112;       // 32*16*512 =   262,144 bf16
  __bf16* xb0    = ws + 1376256;       // 256*1024 bf16 state mirror (ping)
  __bf16* xb1    = ws + 1638400;       // 256*1024 bf16 state mirror (pong)
  __bf16* xhist  = ws + 1900544;       // optional [B,T,X] bf16 history (128 MB)

  const size_t HIST_ELEMS = (size_t)BATCH * STEPS * DIM_X;    // 67,108,864
  bool use_hist = ws_size >= (1900544 + HIST_ELEMS) * sizeof(__bf16);

  pack_kernel<<<336, 256, 0, stream>>>(W, Bs, Wy, ws);

  int stride0 = use_hist ? STEPS * DIM_X : DIM_X;
  __bf16* bf_init = use_hist ? xhist : xb0;
  init_x_kernel<<<1024, 256, 0, stream>>>(x0, x_seq, bf_init, stride0);

  for (int t = 0; t < STEPS - 1; ++t) {
    __bf16* rd = use_hist ? (xhist + (size_t)t * DIM_X)
                          : ((t & 1) ? xb1 : xb0);
    __bf16* wr = use_hist ? (xhist + (size_t)(t + 1) * DIM_X)
                          : ((t & 1) ? xb0 : xb1);
    step_kernel<<<16, 256, 0, stream>>>(rd, wr, stride0, s_seq, u_seq, b,
                                        packW, packBs, x_seq, t);
  }

  if (use_hist)
    y_kernel<true><<<1024, 256, 0, stream>>>(x_seq, xhist, packWy, by, y_seq);
  else
    y_kernel<false><<<1024, 256, 0, stream>>>(x_seq, xhist, packWy, by, y_seq);
}
